// ChannelFusedCrossAttn_2491081031776
// MI455X (gfx1250) — compile-verified, matrix operand order checked by reading, not run on hardware
//
#include <hip/hip_runtime.h>

typedef __attribute__((ext_vector_type(16))) _Float16 v16h;
typedef __attribute__((ext_vector_type(8)))  _Float16 v8h;
typedef __attribute__((ext_vector_type(8)))  float    v8f;

#define BATCH 4
#define NPOS  4096
#define CQ    256
#define CKV   128
#define CIN   512
#define DQK   32
#define NCHUNK (NPOS / 32)

// Tensor Data Mover availability (device pass only; host pass falls back)
#if defined(__gfx1250__) && defined(__has_builtin)
#if __has_builtin(__builtin_amdgcn_tensor_load_to_lds)
#define HAS_TDM 1
#endif
#endif
#ifndef HAS_TDM
#define HAS_TDM 0
#endif

// k-index of element e (0..15) of a 16-bit A/B WMMA fragment, per CDNA5 ISA
// 16-bit 16x32 layout: lane owns row/col (lane&15); hl = lane>>4.
__device__ __forceinline__ int fk(int e, int hl) {
  return ((e >> 3) << 4) + (hl << 3) + (e & 7);
}

// Load a fragment whose 32 k-elements are contiguous within a row:
// two 16B loads at rowbase + hl*8 and rowbase + 16 + hl*8.
__device__ __forceinline__ v16h load_frag_row(const _Float16* rowbase, int hl) {
  union { v16h v; v8h h[2]; } u;
  u.h[0] = *(const v8h*)(rowbase + hl * 8);
  u.h[1] = *(const v8h*)(rowbase + 16 + hl * 8);
  return u.v;
}

// Strided gather for a B fragment from a row-major [K][N] f16 weight matrix.
__device__ __forceinline__ v16h gather_fragB(const _Float16* colbase, int stride, int hl) {
  v16h v;
#pragma unroll
  for (int e = 0; e < 16; ++e) v[e] = colbase[fk(e, hl) * stride];
  return v;
}

// Strided gather + f32->f16 convert for an A fragment from channel-major [C][N] input.
__device__ __forceinline__ v16h gather_fragA_f32(const float* base, int stride, int hl) {
  v16h v;
#pragma unroll
  for (int e = 0; e < 16; ++e) v[e] = (_Float16)base[fk(e, hl) * stride];
  return v;
}

__device__ __forceinline__ v8f wmma16(v16h a, v16h b, v8f c) {
  return __builtin_amdgcn_wmma_f32_16x16x32_f16(false, a, false, b, (short)0, c, false, false);
}

#if HAS_TDM
typedef __attribute__((ext_vector_type(4))) unsigned int tdm_u32x4;
typedef __attribute__((ext_vector_type(8))) int          tdm_i32x8;
typedef __attribute__((ext_vector_type(4))) int          tdm_i32x4;

// Issue a TDM load of a 2-D tile (8-byte elements) from global memory to LDS.
// tile_d1 == 0 -> 1-D contiguous copy of tile_d0 8-byte elements.
// D# bit layout per CDNA5 ISA (groups 0 and 1;
// groups 2/3 zero => tensor up to 2-D).
__device__ __forceinline__ void tdm_load_to_lds(unsigned lds_off, const void* gsrc,
                                                unsigned tile_d0, unsigned tile_d1,
                                                unsigned tens_d0, unsigned tens_d1,
                                                unsigned long long stride0) {
  unsigned long long ga = (unsigned long long)gsrc;
  tdm_u32x4 g0;
  g0.x = 1u;                                                // count=1, user-mode D#
  g0.y = lds_off;                                           // lds_addr (bytes)
  g0.z = (unsigned)ga;                                      // global_addr[31:0]
  g0.w = (unsigned)((ga >> 32) & 0x01FFFFFFu) | (2u << 30); // global_addr[56:32], type=2
  tdm_i32x8 g1;
  g1[0] = (int)(3u << 16);                                  // wg_mask=0, data_size=3 (8B)
  g1[1] = (int)((tens_d0 & 0xFFFFu) << 16);                 // tensor_dim0[15:0]
  g1[2] = (int)((tens_d0 >> 16) & 0xFFFFu) | (int)((tens_d1 & 0xFFFFu) << 16);
  g1[3] = (int)((tens_d1 >> 16) & 0xFFFFu) | (int)(tile_d0 << 16);  // tile_dim0
  g1[4] = (int)(tile_d1 & 0xFFFFu);                         // tile_dim1 (0 => 1-D)
  g1[5] = (int)(stride0 & 0xFFFFFFFFull);                   // tensor_dim0_stride lo
  g1[6] = (int)((stride0 >> 32) & 0xFFFFull);               // stride hi, dim1_stride lo=0
  g1[7] = 0;
  tdm_i32x4 gz = {0, 0, 0, 0};
#if __clang_major__ >= 23
  tdm_i32x8 gz8 = {0, 0, 0, 0, 0, 0, 0, 0};
  __builtin_amdgcn_tensor_load_to_lds(g0, g1, gz, gz, gz8, 0);
#else
  __builtin_amdgcn_tensor_load_to_lds(g0, g1, gz, gz, 0);
#endif
}
#endif  // HAS_TDM

// ---------------------------------------------------------------------------
// Kernel 1: transpose + fp16-convert all weight matrices into [K][N] layouts.
// ---------------------------------------------------------------------------
__global__ __launch_bounds__(256) void prep_weights(
    const float* __restrict__ Wf, const float* __restrict__ Wq,
    const float* __restrict__ Wk, const float* __restrict__ Wv,
    const float* __restrict__ Wo,
    _Float16* __restrict__ WfT, _Float16* __restrict__ WqT,
    _Float16* __restrict__ WkT, _Float16* __restrict__ WvT,
    _Float16* __restrict__ WoT) {
  int i = blockIdx.x * blockDim.x + threadIdx.x;
  if (i < 65536) {                      // WfT [512][128]
    int c = i / 128, o = i % 128;
    WfT[i] = (_Float16)Wf[o * 512 + c];
  } else if (i < 73728) {               // WqT [256][32]
    int j = i - 65536; int c = j / 32, o = j % 32;
    WqT[j] = (_Float16)Wq[o * 256 + c];
  } else if (i < 77824) {               // WkT [128][32]
    int j = i - 73728; int c = j / 32, o = j % 32;
    WkT[j] = (_Float16)Wk[o * 128 + c];
  } else if (i < 110592) {              // WvT [128][256]
    int j = i - 77824; int c = j / 256, o = j % 256;
    WvT[j] = (_Float16)Wv[o * 128 + c];
  } else if (i < 176128) {              // WoT [256][256]
    int j = i - 110592; int c = j / 256, o = j % 256;
    WoT[j] = (_Float16)Wo[o * 256 + c];
  }
}

// ---------------------------------------------------------------------------
// Kernel 2: ctx = LeakyReLU(Wf @ ctx_in + bf); k = Wk@ctx+bk; v = Wv@ctx+bv.
// One wave per 16-position tile. kf: [B][N][32] pos-major; vf: [B][256][N].
// ---------------------------------------------------------------------------
__global__ __launch_bounds__(128) void fuse_kv_kernel(
    const float* __restrict__ context, const _Float16* __restrict__ WfT,
    const float* __restrict__ bf, const _Float16* __restrict__ WkT,
    const float* __restrict__ bk, const _Float16* __restrict__ WvT,
    const float* __restrict__ bv, _Float16* __restrict__ kf,
    _Float16* __restrict__ vf) {
  __shared__ __align__(16) _Float16 ctxs[4][16 * CKV];
  int lane = threadIdx.x & 31;
  int wave = threadIdx.x >> 5;
  int tile = blockIdx.x * 4 + wave;
  int b = tile >> 8;
  int pos0 = (tile & 255) << 4;
  int hl = lane >> 4;
  int col = lane & 15;

  __builtin_prefetch(WfT, 0, 3);   // -> global_prefetch_b8
  __builtin_prefetch(WvT, 0, 3);

  const float* cb = context + (size_t)b * CIN * NPOS + pos0 + col;

  // fusion GEMM: [16 pos x 512] x [512 x 128] -> 8 out-tiles of 16
  v8f co[8];
#pragma unroll
  for (int j = 0; j < 8; ++j) co[j] = (v8f)0.0f;
  for (int t = 0; t < 16; ++t) {
    v16h a = gather_fragA_f32(cb + (size_t)t * 32 * NPOS, NPOS, hl);
#pragma unroll
    for (int j = 0; j < 8; ++j) {
      v16h bb = gather_fragB(WfT + (t * 32) * CKV + j * 16 + col, CKV, hl);
      co[j] = wmma16(a, bb, co[j]);
    }
  }
#pragma unroll
  for (int j = 0; j < 8; ++j) {
    float bias = bf[j * 16 + col];
#pragma unroll
    for (int r = 0; r < 8; ++r) {
      float v = co[j][r] + bias;
      v = v >= 0.0f ? v : 0.1f * v;                 // LeakyReLU(0.1)
      ctxs[wave][(r + 8 * hl) * CKV + j * 16 + col] = (_Float16)v;
    }
  }

  // K projection: [16 x 128] x [128 x 32]
  v8f ka[2];
  ka[0] = (v8f)0.0f; ka[1] = (v8f)0.0f;
#pragma unroll
  for (int t = 0; t < 4; ++t) {
    v16h a = load_frag_row(&ctxs[wave][col * CKV + t * 32], hl);
#pragma unroll
    for (int j = 0; j < 2; ++j) {
      v16h bb = gather_fragB(WkT + (t * 32) * DQK + j * 16 + col, DQK, hl);
      ka[j] = wmma16(a, bb, ka[j]);
    }
  }
#pragma unroll
  for (int j = 0; j < 2; ++j) {
    float bias = bk[j * 16 + col];
#pragma unroll
    for (int r = 0; r < 8; ++r)
      kf[((size_t)b * NPOS + pos0 + r + 8 * hl) * DQK + j * 16 + col] =
          (_Float16)(ka[j][r] + bias);
  }

  // V projection: [16 x 128] x [128 x 256], 4 groups of 4 out-tiles
#pragma unroll 1
  for (int g = 0; g < 4; ++g) {
    v8f va[4];
#pragma unroll
    for (int j = 0; j < 4; ++j) va[j] = (v8f)0.0f;
#pragma unroll
    for (int t = 0; t < 4; ++t) {
      v16h a = load_frag_row(&ctxs[wave][col * CKV + t * 32], hl);
#pragma unroll
      for (int j = 0; j < 4; ++j) {
        int o0 = (g * 4 + j) * 16;
        v16h bb = gather_fragB(WvT + (t * 32) * CQ + o0 + col, CQ, hl);
        va[j] = wmma16(a, bb, va[j]);
      }
    }
#pragma unroll
    for (int j = 0; j < 4; ++j) {
      int o = (g * 4 + j) * 16 + col;
      float bias = bv[o];
#pragma unroll
      for (int r = 0; r < 8; ++r)
        vf[((size_t)b * CQ + o) * NPOS + pos0 + r + 8 * hl] =
            (_Float16)(va[j][r] + bias);
    }
  }
}

// ---------------------------------------------------------------------------
// Kernel 3: q = (Wq @ x + bq) * rsqrt(32). qf: [B][N][32] pos-major.
// ---------------------------------------------------------------------------
__global__ __launch_bounds__(128) void qproj_kernel(
    const float* __restrict__ x, const _Float16* __restrict__ WqT,
    const float* __restrict__ bq, _Float16* __restrict__ qf) {
  int lane = threadIdx.x & 31;
  int wave = threadIdx.x >> 5;
  int tile = blockIdx.x * 4 + wave;
  int b = tile >> 8;
  int pos0 = (tile & 255) << 4;
  int hl = lane >> 4;
  int col = lane & 15;

  __builtin_prefetch(WqT, 0, 3);

  const float* xb = x + (size_t)b * CQ * NPOS + pos0 + col;
  v8f qa[2];
  qa[0] = (v8f)0.0f; qa[1] = (v8f)0.0f;
  for (int t = 0; t < 8; ++t) {
    v16h a = gather_fragA_f32(xb + (size_t)t * 32 * NPOS, NPOS, hl);
#pragma unroll
    for (int j = 0; j < 2; ++j) {
      v16h bb = gather_fragB(WqT + (t * 32) * DQK + j * 16 + col, DQK, hl);
      qa[j] = wmma16(a, bb, qa[j]);
    }
  }
  const float scale = 0.17677669529663687f;        // 32^-0.5 folded into q
#pragma unroll
  for (int j = 0; j < 2; ++j) {
    float bias = bq[j * 16 + col];
#pragma unroll
    for (int r = 0; r < 8; ++r)
      qf[((size_t)b * NPOS + pos0 + r + 8 * hl) * DQK + j * 16 + col] =
          (_Float16)((qa[j][r] + bias) * scale);
  }
}

// ---------------------------------------------------------------------------
// Kernel 4: flash attention with a double-buffered TDM pipeline.
// One wave per 16-query tile. Wave 0 issues the TDM pair for chunk kc+1 into
// the alternate LDS buffer before computing chunk kc; since TENSORcnt retires
// in order, s_wait_tensorcnt 2 after the new issue guarantees the current
// chunk's data has landed, overlapping DMA with 18 WMMAs per chunk.
// hout: [B][N][256] pos-major.
// ---------------------------------------------------------------------------
__global__ __launch_bounds__(128) void attn_kernel(
    const _Float16* __restrict__ qf, const _Float16* __restrict__ kfp,
    const _Float16* __restrict__ vfp, _Float16* __restrict__ hout) {
  __shared__ __align__(16) _Float16 ldsK[2][32 * 32];     // [buf][key][d]
  __shared__ __align__(16) _Float16 ldsV[2][256 * 32];    // [buf][chan][key]
  __shared__ __align__(16) _Float16 pbuf[4][16 * 32];     // per-wave P transpose
  int lane = threadIdx.x & 31;
  int wave = threadIdx.x >> 5;
  int tile = blockIdx.x * 4 + wave;
  int b = tile >> 8;
  int q0 = (tile & 255) << 4;
  int hl = lane >> 4;
  int col = lane & 15;

  v16h qfrag = load_frag_row(qf + ((size_t)b * NPOS + q0 + col) * DQK, hl);

  v8f acc[16];
#pragma unroll
  for (int c = 0; c < 16; ++c) acc[c] = (v8f)0.0f;
  float mrun[8], srun[8];
#pragma unroll
  for (int r = 0; r < 8; ++r) { mrun[r] = -1e30f; srun[r] = 0.0f; }

#if HAS_TDM
  // Prologue: prefill buffer 0 with chunk 0 (K: 2048B contiguous; V: 256x64B).
  if (wave == 0) {
    tdm_load_to_lds((unsigned)(size_t)(void*)ldsK[0],
                    kfp + (size_t)b * NPOS * DQK,
                    256, 0, 1u << 20, 0, 1u << 20);
    tdm_load_to_lds((unsigned)(size_t)(void*)ldsV[0],
                    vfp + (size_t)b * CQ * NPOS,
                    8, 256, 1024, 256, 1024);
  }
#endif

  int cur = 0;
  for (int kc = 0; kc < NCHUNK; ++kc) {
    __syncthreads();   // everyone done reading the buffer about to be refilled
#if HAS_TDM
    if (wave == 0) {
      if (kc + 1 < NCHUNK) {
        // issue next chunk into the alternate buffer, then wait for current
        tdm_load_to_lds((unsigned)(size_t)(void*)ldsK[cur ^ 1],
                        kfp + ((size_t)b * NPOS + (kc + 1) * 32) * DQK,
                        256, 0, 1u << 20, 0, 1u << 20);
        tdm_load_to_lds((unsigned)(size_t)(void*)ldsV[cur ^ 1],
                        vfp + (size_t)b * CQ * NPOS + (kc + 1) * 32,
                        8, 256, 1024, 256, 1024);
        __builtin_amdgcn_s_wait_tensorcnt((short)2);
      } else {
        __builtin_amdgcn_s_wait_tensorcnt((short)0);
      }
    }
#else
    ((float4*)ldsK[cur])[threadIdx.x] =
        ((const float4*)(kfp + ((size_t)b * NPOS + kc * 32) * DQK))[threadIdx.x];
#pragma unroll
    for (int i = 0; i < 8; ++i) {
      int idx = threadIdx.x * 8 + i;
      int c = idx >> 2;
      int m4 = idx & 3;
      ((float4*)ldsV[cur])[idx] =
          *(const float4*)(vfp + ((size_t)b * CQ + c) * NPOS + kc * 32 + m4 * 8);
    }
#endif
    __syncthreads();   // current buffer published to all waves

    // scores for 32 keys: two 16x16x32 WMMAs (K dim = d = 32)
    v16h kf0 = load_frag_row(ldsK[cur] + col * 32, hl);
    v16h kf1 = load_frag_row(ldsK[cur] + (16 + col) * 32, hl);
    v8f s0 = wmma16(qfrag, kf0, (v8f)0.0f);
    v8f s1 = wmma16(qfrag, kf1, (v8f)0.0f);

    // online softmax: row r of the C tile lives at VGPR r, lanes of one half
    float corr[8];
#pragma unroll
    for (int r = 0; r < 8; ++r) {
      float mx = fmaxf(s0[r], s1[r]);
      mx = fmaxf(mx, __shfl_xor(mx, 1, 32));
      mx = fmaxf(mx, __shfl_xor(mx, 2, 32));
      mx = fmaxf(mx, __shfl_xor(mx, 4, 32));
      mx = fmaxf(mx, __shfl_xor(mx, 8, 32));
      float mn = fmaxf(mrun[r], mx);
      corr[r] = __expf(mrun[r] - mn);
      mrun[r] = mn;
      float p0 = __expf(s0[r] - mn);
      float p1 = __expf(s1[r] - mn);
      float ps = p0 + p1;
      ps += __shfl_xor(ps, 1, 32);
      ps += __shfl_xor(ps, 2, 32);
      ps += __shfl_xor(ps, 4, 32);
      ps += __shfl_xor(ps, 8, 32);
      srun[r] = srun[r] * corr[r] + ps;
      // transpose P into A-fragment layout via per-wave LDS scratch
      pbuf[wave][(r + 8 * hl) * 32 + col] = (_Float16)p0;
      pbuf[wave][(r + 8 * hl) * 32 + 16 + col] = (_Float16)p1;
    }
#pragma unroll
    for (int c = 0; c < 16; ++c)
#pragma unroll
      for (int r = 0; r < 8; ++r) acc[c][r] *= corr[r];

    // accumulate h += P[16x32] x V[32x16] for 16 channel tiles
    v16h pfrag = load_frag_row(&pbuf[wave][col * 32], hl);
#pragma unroll
    for (int c = 0; c < 16; ++c) {
      v16h vfrag = load_frag_row(ldsV[cur] + (c * 16 + col) * 32, hl);
      acc[c] = wmma16(pfrag, vfrag, acc[c]);
    }
    cur ^= 1;
  }

  float inv[8];
#pragma unroll
  for (int r = 0; r < 8; ++r) inv[r] = 1.0f / srun[r];
#pragma unroll
  for (int c = 0; c < 16; ++c)
#pragma unroll
    for (int r = 0; r < 8; ++r)
      hout[((size_t)b * NPOS + q0 + r + 8 * hl) * CQ + c * 16 + col] =
          (_Float16)(acc[c][r] * inv[r]);
}

// ---------------------------------------------------------------------------
// Kernel 5: out = gamma * (Wo @ h + bo) + x  (fp32 output, channel-major)
// ---------------------------------------------------------------------------
__global__ __launch_bounds__(128) void outproj_kernel(
    const _Float16* __restrict__ hbuf, const _Float16* __restrict__ WoT,
    const float* __restrict__ bo, const float* __restrict__ x,
    const float* __restrict__ gamma, float* __restrict__ out) {
  int lane = threadIdx.x & 31;
  int wave = threadIdx.x >> 5;
  int tile = blockIdx.x * 4 + wave;
  int b = tile >> 8;
  int pos0 = (tile & 255) << 4;
  int hl = lane >> 4;
  int col = lane & 15;

  __builtin_prefetch(WoT, 0, 3);

  const _Float16* hb = hbuf + ((size_t)b * NPOS + pos0 + col) * CQ;
  float g = gamma[0];

#pragma unroll 1
  for (int gi = 0; gi < 4; ++gi) {
    v8f oa[4];
#pragma unroll
    for (int j = 0; j < 4; ++j) oa[j] = (v8f)0.0f;
#pragma unroll 1
    for (int t = 0; t < 8; ++t) {
      v16h a = load_frag_row(hb + t * 32, hl);
#pragma unroll
      for (int j = 0; j < 4; ++j) {
        v16h bb = gather_fragB(WoT + (t * 32) * CQ + (gi * 4 + j) * 16 + col, CQ, hl);
        oa[j] = wmma16(a, bb, oa[j]);
      }
    }
#pragma unroll
    for (int j = 0; j < 4; ++j) {
      int o = (gi * 4 + j) * 16 + col;
      float bias = bo[o];
#pragma unroll
      for (int r = 0; r < 8; ++r) {
        size_t xi = ((size_t)b * CQ + o) * NPOS + pos0 + r + 8 * hl;
        out[xi] = g * (oa[j][r] + bias) + x[xi];
      }
    }
  }
}

// ---------------------------------------------------------------------------
extern "C" void kernel_launch(void* const* d_in, const int* in_sizes, int n_in,
                              void* d_out, int out_size, void* d_ws, size_t ws_size,
                              hipStream_t stream) {
  const float* x     = (const float*)d_in[0];
  const float* ctx   = (const float*)d_in[1];
  const float* Wf    = (const float*)d_in[2];
  const float* bf    = (const float*)d_in[3];
  const float* Wq    = (const float*)d_in[4];
  const float* bq    = (const float*)d_in[5];
  const float* Wk    = (const float*)d_in[6];
  const float* bk    = (const float*)d_in[7];
  const float* Wv    = (const float*)d_in[8];
  const float* bv    = (const float*)d_in[9];
  const float* Wo    = (const float*)d_in[10];
  const float* bo    = (const float*)d_in[11];
  const float* gamma = (const float*)d_in[12];
  float* out = (float*)d_out;

  char* ws = (char*)d_ws;
  _Float16* WfT = (_Float16*)(ws + 0);          // 512*128*2  = 131072
  _Float16* WqT = (_Float16*)(ws + 131072);     // 256*32*2   =  16384
  _Float16* WkT = (_Float16*)(ws + 147456);     // 128*32*2   =   8192
  _Float16* WvT = (_Float16*)(ws + 155648);     // 128*256*2  =  65536
  _Float16* WoT = (_Float16*)(ws + 221184);     // 256*256*2  = 131072
  _Float16* qf  = (_Float16*)(ws + 352256);     // B*N*32*2   = 1048576
  _Float16* kf  = (_Float16*)(ws + 1400832);    // B*N*32*2   = 1048576
  _Float16* vf  = (_Float16*)(ws + 2449408);    // B*256*N*2  = 8388608
  _Float16* hb  = (_Float16*)(ws + 10838016);   // B*N*256*2  = 8388608

  prep_weights<<<688, 256, 0, stream>>>(Wf, Wq, Wk, Wv, Wo, WfT, WqT, WkT, WvT, WoT);
  fuse_kv_kernel<<<256, 128, 0, stream>>>(ctx, WfT, bf, WkT, bk, WvT, bv, kf, vf);
  qproj_kernel<<<256, 128, 0, stream>>>(x, WqT, bq, qf);
  attn_kernel<<<256, 128, 0, stream>>>(qf, kf, vf, hb);
  outproj_kernel<<<256, 128, 0, stream>>>(hb, WoT, bo, x, gamma, out);
}
